// Multihead_Self_Attention_20332375180127
// MI455X (gfx1250) — matrix operand order repaired and checked by reading, not compiled, not run on hardware
//
#include <hip/hip_runtime.h>

#define D_MODEL 1024
#define NUM_HEADS 16
#define D_K 64
#define SEQ 2048
#define BATCH 2
#define MROWS (BATCH * SEQ) /* 4096 */

typedef __attribute__((ext_vector_type(4)))  float    v4f;
typedef __attribute__((ext_vector_type(4)))  _Float16 v4h;
typedef __attribute__((ext_vector_type(8)))  _Float16 v8h;
typedef __attribute__((ext_vector_type(16))) _Float16 v16h;
typedef __attribute__((ext_vector_type(8)))  float    v8f;

union V16U { v16h v; v8h h[2]; };

// scale folded into Q at projection time: 1/sqrt(64) * log2(e)
#define Q_SCALE 0.18033688011112042f

#if __has_builtin(__builtin_amdgcn_exp2f)
#define EXP2(x) __builtin_amdgcn_exp2f(x)   // raw v_exp_f32, no denorm fixup
#else
#define EXP2(x) exp2f(x)
#endif

__device__ __forceinline__ v8f wmma16(const V16U& a, const V16U& b, v8f c) {
  // D = A(16x32 f16) * B(32x16 f16) + C(16x16 f32)
  return __builtin_amdgcn_wmma_f32_16x16x32_f16(
      /*neg_a=*/false, a.v, /*neg_b=*/false, b.v,
      /*c_mod=*/(short)0, c, /*reuse_a=*/false, /*reuse_b=*/false);
}

// ---------------------------------------------------------------- converts (x4 vectorized)
__global__ void cvt_f32_to_f16_v4(const v4f* __restrict__ src,
                                  v4h* __restrict__ dst, int n4) {
  int i = blockIdx.x * blockDim.x + threadIdx.x;
  if (i < n4) {
    v4f x = src[i];
    v4h y;
    y[0] = (_Float16)x[0]; y[1] = (_Float16)x[1];
    y[2] = (_Float16)x[2]; y[3] = (_Float16)x[3];
    dst[i] = y;
  }
}

// ---------------------------------------------------------------- projection GEMM
// out[m,n] = sum_k A[m,k] * W[n,k]; each wave computes a 32(M) x 64(N) tile:
// 8 WMMA per 32-wide k-step fed by 12 b128 loads (1.5 loads/WMMA).
// MODE 0: Q (+RoPE, *Q_SCALE) -> out16 as [b][h][s][64]
// MODE 1: K (+RoPE)           -> out16 as [b][h][s][64]
// MODE 2: V                   -> out16 as [b][h][64][s]  (transposed, key-contiguous)
// MODE 3: final               -> out32 as [m][1024] fp32
template <int MODE>
__global__ __launch_bounds__(256) void proj_gemm(
    const _Float16* __restrict__ A, const _Float16* __restrict__ W,
    _Float16* __restrict__ out16, float* __restrict__ out32,
    const int* __restrict__ pos) {
  const int wid  = threadIdx.x >> 5;
  const int lane = threadIdx.x & 31;
  const int lm   = lane & 15;
  const int hi   = lane >> 4;

  const int gw   = blockIdx.x * 8 + wid;   // 2048 wave tiles
  const int mt   = gw >> 4;                // 128 row groups of 32
  const int nt   = gw & 15;                // 16  col groups of 64
  const int row0 = mt * 32, col0 = nt * 64;

  const int aoff = hi ? 8 : 0;   // A-fragment lane chunk offset
  const int boff = hi ? 8 : 0;   // B-fragment lane chunk offset

  const _Float16* arow0 = A + (size_t)(row0 + lm) * D_MODEL;
  const _Float16* arow1 = arow0 + (size_t)16 * D_MODEL;

  v8f acc[2][4] = {};
  for (int kb = 0; kb < D_MODEL; kb += 32) {
    V16U a0, a1;
    a0.h[0] = *(const v8h*)(arow0 + kb + aoff);
    a0.h[1] = *(const v8h*)(arow0 + kb + 16 + aoff);
    a1.h[0] = *(const v8h*)(arow1 + kb + aoff);
    a1.h[1] = *(const v8h*)(arow1 + kb + 16 + aoff);
#pragma unroll
    for (int j = 0; j < 4; ++j) {
      const _Float16* brow = W + (size_t)(col0 + j * 16 + lm) * D_MODEL + kb;
      V16U b;
      b.h[0] = *(const v8h*)(brow + boff);
      b.h[1] = *(const v8h*)(brow + boff + 16);
      acc[0][j] = wmma16(a0, b, acc[0][j]);
      acc[1][j] = wmma16(a1, b, acc[1][j]);
    }
  }

#pragma unroll
  for (int j = 0; j < 4; ++j) {
    const int col  = col0 + j * 16 + lm;  // global column 0..1023
    const int head = col >> 6;
    const int d    = col & 63;
    // inv_freq = theta^(-(d & ~1)/64), ln(10000) = 9.2103403719761836
    const float inv_freq = __expf(-(float)(d & ~1) * (9.2103403719761836f / 64.0f));
#pragma unroll
    for (int mi = 0; mi < 2; ++mi) {
#pragma unroll
      for (int r = 0; r < 8; ++r) {
        const int row = row0 + mi * 16 + r + 8 * hi;  // global row (b*SEQ + s)
        float v = acc[mi][j][r];
        if (MODE <= 1) {
          const int b = row >> 11, s = row & (SEQ - 1);
          const float p = (float)pos[s];
          float sn, cs;
          __sincosf(p * inv_freq, &sn, &cs);
          const float other = __shfl_xor(v, 1, 32);  // RoPE pair: adjacent column/lane
          float o = (d & 1) ? (other * sn + v * cs) : (v * cs - other * sn);
          if (MODE == 0) o *= Q_SCALE;
          out16[((size_t)(b * NUM_HEADS + head) * SEQ + s) * D_K + d] = (_Float16)o;
        } else if (MODE == 2) {
          const int b = row >> 11, s = row & (SEQ - 1);
          out16[((size_t)(b * NUM_HEADS + head) * D_K + d) * SEQ + s] = (_Float16)v;
        } else {
          out32[(size_t)row * D_MODEL + col] = v;
        }
      }
    }
  }
}

// ---------------------------------------------------------------- flash attention
// Transposed formulation: S^T = K*Q^T, O^T = V^T * P^T.
// Lanes index the query -> softmax stats are per-lane scalars; key reductions are
// in-register + a single shfl_xor(16). One wave owns a 16-query tile of one (b,h).
__global__ __launch_bounds__(256) void attn_kernel(
    const _Float16* __restrict__ Qh,  // [b][h][s][64]  (pre-scaled by Q_SCALE)
    const _Float16* __restrict__ Kh,  // [b][h][s][64]
    const _Float16* __restrict__ Vt,  // [b][h][64][s]
    _Float16* __restrict__ Oh) {      // [b][s][1024]
  __shared__ __align__(16) _Float16 pbuf[8][16 * 40];  // per-wave P^T tile [q][key], pitch 40

  const int wid  = threadIdx.x >> 5;
  const int lane = threadIdx.x & 31;
  const int lm   = lane & 15;
  const int hi   = lane >> 4;

  const int gw = blockIdx.x * 8 + wid;  // 4096 wave tiles
  const int qt = gw & 127;              // query tile in [0,128)
  const int bh = gw >> 7;               // b*16 + h

  const int aoff = hi ? 8 : 0;
  const int boff = hi ? 8 : 0;

  // Q^T B-fragments (d-chunks 0..31 and 32..63), resident for the whole loop
  const _Float16* qrow = Qh + ((size_t)bh * SEQ + qt * 16 + lm) * D_K;
  V16U bq0, bq1;
  bq0.h[0] = *(const v8h*)(qrow + boff);
  bq0.h[1] = *(const v8h*)(qrow + boff + 16);
  bq1.h[0] = *(const v8h*)(qrow + 32 + boff);
  bq1.h[1] = *(const v8h*)(qrow + 32 + boff + 16);

  v8f o0 = {}, o1 = {}, o2 = {}, o3 = {};  // O^T d-tiles (d x q), accumulators
  float mrun = -1e30f, lrun = 0.0f;        // per-lane (per-query) running stats

  const _Float16* kbh = Kh + (size_t)bh * SEQ * D_K;
  const _Float16* vbh = Vt + (size_t)bh * D_K * SEQ;
  _Float16* lds = &pbuf[wid][0];
  const int q = qt * 16 + lm;

  auto process = [&](int kbase, bool domask) {
    // ---- scores S^T: two 16(key) x 16(q) tiles, K as A-fragment, Q^T as B
    v8f s0 = {}, s1 = {};
    {
      const _Float16* k0 = kbh + (size_t)(kbase + lm) * D_K;
      const _Float16* k1 = k0 + (size_t)16 * D_K;
      V16U a;
      a.h[0] = *(const v8h*)(k0 + aoff);      a.h[1] = *(const v8h*)(k0 + 16 + aoff);
      s0 = wmma16(a, bq0, s0);
      a.h[0] = *(const v8h*)(k0 + 32 + aoff); a.h[1] = *(const v8h*)(k0 + 48 + aoff);
      s0 = wmma16(a, bq1, s0);
      a.h[0] = *(const v8h*)(k1 + aoff);      a.h[1] = *(const v8h*)(k1 + 16 + aoff);
      s1 = wmma16(a, bq0, s1);
      a.h[0] = *(const v8h*)(k1 + 32 + aoff); a.h[1] = *(const v8h*)(k1 + 48 + aoff);
      s1 = wmma16(a, bq1, s1);
    }
    // ---- causal mask: diagonal block only. key index = kbase + {0,16} + r + 8*hi
    if (domask) {
#pragma unroll
      for (int r = 0; r < 8; ++r) {
        s0[r] = (kbase + r + 8 * hi      <= q) ? s0[r] : -1e30f;
        s1[r] = (kbase + 16 + r + 8 * hi <= q) ? s1[r] : -1e30f;
      }
    }
    // ---- online softmax in exp2 domain (scale pre-folded into Q)
    float t = s0[0];
#pragma unroll
    for (int r = 1; r < 8; ++r) t = fmaxf(t, s0[r]);
#pragma unroll
    for (int r = 0; r < 8; ++r) t = fmaxf(t, s1[r]);
    t = fmaxf(t, __shfl_xor(t, 16, 32));   // combine key halves (lanes share q)
    const float mnew  = fmaxf(mrun, t);
    const float alpha = EXP2(mrun - mnew);
    mrun = mnew;
    v8h ph0, ph1;
    float rs = 0.0f;
#pragma unroll
    for (int r = 0; r < 8; ++r) {
      const float p0 = EXP2(s0[r] - mnew);
      const float p1 = EXP2(s1[r] - mnew);
      rs += p0 + p1;
      ph0[r] = (_Float16)p0;
      ph1[r] = (_Float16)p1;
    }
    rs += __shfl_xor(rs, 16, 32);
    lrun = lrun * alpha + rs;
#pragma unroll
    for (int r = 0; r < 8; ++r) {
      o0[r] *= alpha; o1[r] *= alpha; o2[r] *= alpha; o3[r] *= alpha;
    }
    // ---- stage P^T [q][key] in LDS: key-consecutive -> two b128 stores/lane
    *(v8h*)(lds + lm * 40 + 8 * hi)      = ph0;
    *(v8h*)(lds + lm * 40 + 16 + 8 * hi) = ph1;
    asm volatile("s_wait_dscnt 0" ::: "memory");
    V16U bp;  // P^T B-fragment: lane = q, slots = key chunk
    bp.h[0] = *(const v8h*)(lds + lm * 40 + boff);
    bp.h[1] = *(const v8h*)(lds + lm * 40 + boff + 16);
    // ---- O^T += V^T * P^T over four 16-wide d tiles (V^T rows are key-contiguous)
    const _Float16* vp = vbh + (size_t)lm * SEQ + kbase;
    V16U av;
    av.h[0] = *(const v8h*)(vp + aoff); av.h[1] = *(const v8h*)(vp + 16 + aoff);
    o0 = wmma16(av, bp, o0); vp += (size_t)16 * SEQ;
    av.h[0] = *(const v8h*)(vp + aoff); av.h[1] = *(const v8h*)(vp + 16 + aoff);
    o1 = wmma16(av, bp, o1); vp += (size_t)16 * SEQ;
    av.h[0] = *(const v8h*)(vp + aoff); av.h[1] = *(const v8h*)(vp + 16 + aoff);
    o2 = wmma16(av, bp, o2); vp += (size_t)16 * SEQ;
    av.h[0] = *(const v8h*)(vp + aoff); av.h[1] = *(const v8h*)(vp + 16 + aoff);
    o3 = wmma16(av, bp, o3);
  };

  const int nfull = qt >> 1;  // full (unmasked) 32-key blocks below the diagonal
  for (int kb = 0; kb < nfull; ++kb) process(kb * 32, false);
  process(nfull * 32, true);  // exactly one diagonal (masked) block

  // ---- finalize: O^T element (d = dt*16 + r + 8*hi, q = lm); d-consecutive -> b128 stores
  const int b = bh >> 4, h = bh & 15;
  const float inv = 1.0f / lrun;
  _Float16* obase = Oh + ((size_t)(b * SEQ + q)) * D_MODEL + h * D_K + 8 * hi;
  v8h oh;
#pragma unroll
  for (int r = 0; r < 8; ++r) oh[r] = (_Float16)(o0[r] * inv);
  *(v8h*)(obase) = oh;
#pragma unroll
  for (int r = 0; r < 8; ++r) oh[r] = (_Float16)(o1[r] * inv);
  *(v8h*)(obase + 16) = oh;
#pragma unroll
  for (int r = 0; r < 8; ++r) oh[r] = (_Float16)(o2[r] * inv);
  *(v8h*)(obase + 32) = oh;
#pragma unroll
  for (int r = 0; r < 8; ++r) oh[r] = (_Float16)(o3[r] * inv);
  *(v8h*)(obase + 48) = oh;
}

// ---------------------------------------------------------------- launch
extern "C" void kernel_launch(void* const* d_in, const int* in_sizes, int n_in,
                              void* d_out, int out_size, void* d_ws, size_t ws_size,
                              hipStream_t stream) {
  (void)in_sizes; (void)n_in; (void)out_size; (void)ws_size;
  const float* X  = (const float*)d_in[0];
  const float* Wq = (const float*)d_in[1];
  const float* Wk = (const float*)d_in[2];
  const float* Wv = (const float*)d_in[3];
  const float* Wo = (const float*)d_in[4];
  const int*  pos = (const int*)d_in[5];

  char* ws = (char*)d_ws;
  size_t off = 0;
  auto alloc = [&](size_t bytes) {
    char* p = ws + off;
    off += (bytes + 255) & ~(size_t)255;
    return p;
  };
  _Float16* Xh  = (_Float16*)alloc((size_t)MROWS * D_MODEL * 2);
  _Float16* Wqh = (_Float16*)alloc((size_t)D_MODEL * D_MODEL * 2);
  _Float16* Wkh = (_Float16*)alloc((size_t)D_MODEL * D_MODEL * 2);
  _Float16* Wvh = (_Float16*)alloc((size_t)D_MODEL * D_MODEL * 2);
  _Float16* Woh = (_Float16*)alloc((size_t)D_MODEL * D_MODEL * 2);
  _Float16* Qh  = (_Float16*)alloc((size_t)MROWS * D_MODEL * 2);
  _Float16* Kh  = (_Float16*)alloc((size_t)MROWS * D_MODEL * 2);
  _Float16* Vt  = (_Float16*)alloc((size_t)MROWS * D_MODEL * 2);
  _Float16* Oh  = (_Float16*)alloc((size_t)MROWS * D_MODEL * 2);

  const int nX4 = MROWS * D_MODEL / 4, nW4 = D_MODEL * D_MODEL / 4;
  cvt_f32_to_f16_v4<<<(nX4 + 255) / 256, 256, 0, stream>>>((const v4f*)X, (v4h*)Xh, nX4);
  cvt_f32_to_f16_v4<<<(nW4 + 255) / 256, 256, 0, stream>>>((const v4f*)Wq, (v4h*)Wqh, nW4);
  cvt_f32_to_f16_v4<<<(nW4 + 255) / 256, 256, 0, stream>>>((const v4f*)Wk, (v4h*)Wkh, nW4);
  cvt_f32_to_f16_v4<<<(nW4 + 255) / 256, 256, 0, stream>>>((const v4f*)Wv, (v4h*)Wvh, nW4);
  cvt_f32_to_f16_v4<<<(nW4 + 255) / 256, 256, 0, stream>>>((const v4f*)Wo, (v4h*)Woh, nW4);

  // 2048 32x64 wave tiles, 8 waves/block -> 256 blocks
  proj_gemm<0><<<256, 256, 0, stream>>>(Xh, Wqh, Qh, nullptr, pos);
  proj_gemm<1><<<256, 256, 0, stream>>>(Xh, Wkh, Kh, nullptr, pos);
  proj_gemm<2><<<256, 256, 0, stream>>>(Xh, Wvh, Vt, nullptr, pos);

  // 2*16*128 = 4096 wave tiles -> 512 blocks
  attn_kernel<<<512, 256, 0, stream>>>(Qh, Kh, Vt, Oh);

  proj_gemm<3><<<256, 256, 0, stream>>>(Oh, Woh, nullptr, (float*)d_out, pos);
}
